// SectorInteractionGraph_26843545600095
// MI455X (gfx1250) — matrix operand order repaired and checked by reading, not compiled, not run on hardware
//
#include <hip/hip_runtime.h>
#include <hip/hip_bf16.h>

#define BATCH  32768
#define SEC    12
#define DIM    128
#define NHEAD  4
#define HDIM   32
#define NLAYER 2
#define NBAT   4      // batches (waves) per workgroup
#define BLOCK  128    // 4 waves of 32
#define XS     132    // fp32 activation row stride (bank-spread, 8B-align pairs)
#define QS     136    // f16 q/k/v row stride (16B-aligned rows, bank-spread)
#define WS     136    // f16 weight row stride
#define PS     32     // prob row stride

typedef _Float16 v16h __attribute__((ext_vector_type(16)));
typedef _Float16 v8h  __attribute__((ext_vector_type(8)));
typedef float    v8f  __attribute__((ext_vector_type(8)));

__device__ __forceinline__ void lds_fence() {
  // wave-private LDS RAW: DS ops are in-order per wave; this pins the
  // compiler ordering and drains the CDNA5 split DS counter.
  __asm__ volatile("s_wait_dscnt 0" ::: "memory");
}

__device__ __forceinline__ v8f wmma16(v16h a, v16h b, v8f c) {
  return __builtin_amdgcn_wmma_f32_16x16x32_f16(false, a, false, b, (short)0, c,
                                                false, false);
}

// A-fragment (16x32 f16): lane holds row m=lane&15; K = 16*(lane>>4) folded into
// the pointer; per-VGPR K offset = 2*v.
__device__ __forceinline__ v16h afrag_f32(const float* p) {
  v16h a;
#pragma unroll
  for (int i = 0; i < 8; ++i) {
    const int k = i << 1;
    a[2*i]   = (_Float16)p[k];
    a[2*i+1] = (_Float16)p[k+1];
  }
  return a;
}
__device__ __forceinline__ v16h afrag_f16(const _Float16* p) {
  v16h a;
#pragma unroll
  for (int i = 0; i < 8; ++i) {
    const int k = i << 1;
    a[2*i]   = p[k];
    a[2*i+1] = p[k+1];
  }
  return a;
}
// B-fragment (32x16 f16): lane = column n; 16 contiguous halves of K
// (K base = 16*(lane>>4) folded into pointer). Two aligned 16B LDS loads.
__device__ __forceinline__ v16h bfrag(const _Float16* p) {
  const v8h* q = (const v8h*)p;
  v8h lo = q[0], hi = q[1];
  v16h b;
#pragma unroll
  for (int i = 0; i < 8; ++i) { b[i] = lo[i]; b[8+i] = hi[i]; }
  return b;
}

// cooperative: W (global fp32 [k][n], 128x128) -> wbuf f16 transposed [n][k]
__device__ __forceinline__ void load_weight(_Float16* wb, const float* __restrict__ W) {
  for (int i = threadIdx.x; i < 128*128; i += BLOCK) {
    const int k = i >> 7, n = i & 127;
    wb[n*WS + k] = (_Float16)W[i];
  }
}

__global__ __launch_bounds__(BLOCK) void sector_graph_kernel(
    const float* __restrict__ sf,    const float* __restrict__ regime,
    const float* __restrict__ ip_w,  const float* __restrict__ ip_b,
    const float* __restrict__ gate_w,const float* __restrict__ gate_b,
    const float* __restrict__ wq,    const float* __restrict__ bq,
    const float* __restrict__ wk,    const float* __restrict__ bk,
    const float* __restrict__ wv,    const float* __restrict__ bv,
    const float* __restrict__ wo,    const float* __restrict__ bo,
    const float* __restrict__ adj,   const float* __restrict__ ln_g,
    const float* __restrict__ ln_b,  float* __restrict__ out)
{
  __shared__ float    xf[NBAT][16*XS];     // fp32 master activations
  __shared__ _Float16 wbuf[128*WS];        // staged weight (transposed, f16)
  __shared__ _Float16 qb[NBAT][16*QS];     // q, later attention-out
  __shared__ _Float16 kb[NBAT][16*QS];
  __shared__ _Float16 vb[NBAT][16*QS];
  __shared__ _Float16 pb[NBAT][16*PS];     // softmax probs (keys padded to 32)

  const int tid   = threadIdx.x;
  const int wid   = tid >> 5;
  const int lane  = tid & 31;
  const int g     = lane >> 4;   // half-wave group
  const int nl    = lane & 15;
  const int batch = blockIdx.x * NBAT + wid;

  float*    X  = xf[wid];
  _Float16* Qs = qb[wid];
  _Float16* Ks = kb[wid];
  _Float16* Vs = vb[wid];
  _Float16* Ps = pb[wid];

  // zero-init LDS: padding rows (12..15) and padded key columns must be 0
  for (int i = tid; i < NBAT*16*XS; i += BLOCK) ((float*)xf)[i] = 0.f;
  for (int i = tid; i < NBAT*16*QS; i += BLOCK) {
    ((_Float16*)qb)[i] = (_Float16)0; ((_Float16*)kb)[i] = (_Float16)0;
    ((_Float16*)vb)[i] = (_Float16)0;
  }
  for (int i = tid; i < NBAT*16*PS; i += BLOCK) ((_Float16*)pb)[i] = (_Float16)0;
  __syncthreads();

  // stage sector_features (coalesced) into X
  {
    const float* sfb = sf + (size_t)batch * SEC * DIM;
    for (int i = lane; i < SEC*DIM; i += 32)
      X[(i >> 7)*XS + (i & 127)] = sfb[i];
  }
  lds_fence();

  v16h af[4];
  v8f  acc[8];

  // ================= input projection: x = sf @ ip_w + ip_b =================
  __syncthreads();
  load_weight(wbuf, ip_w);
  __syncthreads();
  {
    const float* xr = X + nl*XS + g*16;
#pragma unroll
    for (int kf = 0; kf < 4; ++kf) af[kf] = afrag_f32(xr + kf*32);
  }
#pragma unroll
  for (int nt = 0; nt < 8; ++nt) {
    v8f c = {};
    const _Float16* wr = wbuf + (nt*16 + nl)*WS + g*16;
#pragma unroll
    for (int kf = 0; kf < 4; ++kf) c = wmma16(af[kf], bfrag(wr + kf*32), c);
    acc[nt] = c;
  }
#pragma unroll
  for (int nt = 0; nt < 8; ++nt) {
    const float bias = ip_b[nt*16 + nl];
#pragma unroll
    for (int r = 0; r < 8; ++r) {
      const int m = r + 8*g;
      if (m < SEC) X[m*XS + nt*16 + nl] = acc[nt][r] + bias;
    }
  }
  lds_fence();

  // ================= regime gate: x *= sigmoid([x|regime] @ gate_w + b) =====
  __syncthreads();
  load_weight(wbuf, gate_w);                 // input dims 0..127 (x half)
  __syncthreads();
  {
    const float* xr = X + nl*XS + g*16;
#pragma unroll
    for (int kf = 0; kf < 4; ++kf) af[kf] = afrag_f32(xr + kf*32);
  }
#pragma unroll
  for (int nt = 0; nt < 8; ++nt) {
    v8f c = {};
    const _Float16* wr = wbuf + (nt*16 + nl)*WS + g*16;
#pragma unroll
    for (int kf = 0; kf < 4; ++kf) c = wmma16(af[kf], bfrag(wr + kf*32), c);
    acc[nt] = c;
  }
  __syncthreads();
  load_weight(wbuf, gate_w + 128*DIM);       // input dims 128..255 (regime half)
  __syncthreads();
  {
    // regime row is identical for every sector m -> same fragment for all rows
    const float* rr = regime + (size_t)batch*DIM + g*16;
#pragma unroll
    for (int kf = 0; kf < 4; ++kf) af[kf] = afrag_f32(rr + kf*32);
  }
#pragma unroll
  for (int nt = 0; nt < 8; ++nt) {
    v8f c = acc[nt];
    const _Float16* wr = wbuf + (nt*16 + nl)*WS + g*16;
#pragma unroll
    for (int kf = 0; kf < 4; ++kf) c = wmma16(af[kf], bfrag(wr + kf*32), c);
    const float gbv = gate_b[nt*16 + nl];
#pragma unroll
    for (int r = 0; r < 8; ++r) {
      const int m = r + 8*g;
      if (m < SEC) {
        const float z  = c[r] + gbv;
        const float gt = 1.f / (1.f + __expf(-z));
        X[m*XS + nt*16 + nl] *= gt;
      }
    }
  }
  lds_fence();

  // ================= transformer layers =================
  const float scl = 0.17677669529663689f;    // 1/sqrt(HD)
  for (int l = 0; l < NLAYER; ++l) {
    v16h xfr[4];
    {
      const float* xr = X + nl*XS + g*16;
#pragma unroll
      for (int kf = 0; kf < 4; ++kf) xfr[kf] = afrag_f32(xr + kf*32);
    }

    // ---- Q, K, V projections (x frags reused across all three) ----
    const float* Wqkv[3] = { wq + (size_t)l*DIM*DIM, wk + (size_t)l*DIM*DIM,
                             wv + (size_t)l*DIM*DIM };
    const float* Bqkv[3] = { bq + l*DIM, bk + l*DIM, bv + l*DIM };
    _Float16*    Dqkv[3] = { Qs, Ks, Vs };
    for (int t = 0; t < 3; ++t) {
      __syncthreads();
      load_weight(wbuf, Wqkv[t]);
      __syncthreads();
#pragma unroll
      for (int nt = 0; nt < 8; ++nt) {
        v8f c = {};
        const _Float16* wr = wbuf + (nt*16 + nl)*WS + g*16;
#pragma unroll
        for (int kf = 0; kf < 4; ++kf) c = wmma16(xfr[kf], bfrag(wr + kf*32), c);
        const float bias = Bqkv[t][nt*16 + nl];
#pragma unroll
        for (int r = 0; r < 8; ++r) {
          const int m = r + 8*g;
          if (m < SEC) Dqkv[t][m*QS + nt*16 + nl] = (_Float16)(c[r] + bias);
        }
      }
    }
    lds_fence();

    // ---- attention per head: scores WMMA, softmax, attn@V WMMA ----
    for (int h = 0; h < NHEAD; ++h) {
      const v16h aq   = afrag_f16(Qs + nl*QS + h*HDIM + g*16);
      const v16h bk16 = bfrag(Ks + nl*QS + h*HDIM + g*16);  // K rows read as B
      v8f sc = {};
      sc = wmma16(aq, bk16, sc);
#pragma unroll
      for (int r = 0; r < 8; ++r) {
        const int m = r + 8*g;              // query sector; nl = key sector
        float s;
        if (nl < SEC && m < SEC)
          s = sc[r]*scl + adj[(((size_t)l*NHEAD + h)*SEC + m)*SEC + nl];
        else
          s = -1e30f;
        float mx = s;
#pragma unroll
        for (int off = 1; off < 16; off <<= 1)
          mx = fmaxf(mx, __shfl_xor(mx, off, 32));
        const float e = (nl < SEC) ? __expf(s - mx) : 0.f;
        float sm = e;
#pragma unroll
        for (int off = 1; off < 16; off <<= 1) sm += __shfl_xor(sm, off, 32);
        if (m < SEC && nl < SEC) Ps[m*PS + nl] = (_Float16)(e / sm);
      }
      lds_fence();
      const v16h ap = afrag_f16(Ps + nl*PS + g*16);   // probs, keys padded to 32
#pragma unroll
      for (int t2 = 0; t2 < 2; ++t2) {
        const int col = h*HDIM + t2*16 + nl;
        v16h bvv;
        if (g == 0) {                     // K (sector) 0..15; 16..31 are zero
#pragma unroll
          for (int j = 0; j < 16; ++j) bvv[j] = Vs[j*QS + col];
        } else {
#pragma unroll
          for (int j = 0; j < 16; ++j) bvv[j] = (_Float16)0;
        }
        v8f oc = {};
        oc = wmma16(ap, bvv, oc);
#pragma unroll
        for (int r = 0; r < 8; ++r) {     // store attn-out over q (cols done)
          const int m = r + 8*g;
          if (m < SEC) Qs[m*QS + col] = (_Float16)oc[r];
        }
      }
      lds_fence();
    }

    // ---- output projection + bias + residual + LayerNorm ----
    __syncthreads();
    load_weight(wbuf, wo + (size_t)l*DIM*DIM);
    __syncthreads();
    v16h ofr[4];
    {
      const _Float16* orow = Qs + nl*QS + g*16;
#pragma unroll
      for (int kf = 0; kf < 4; ++kf) ofr[kf] = afrag_f16(orow + kf*32);
    }
#pragma unroll
    for (int nt = 0; nt < 8; ++nt) {
      v8f c = {};
      const _Float16* wr = wbuf + (nt*16 + nl)*WS + g*16;
#pragma unroll
      for (int kf = 0; kf < 4; ++kf) c = wmma16(ofr[kf], bfrag(wr + kf*32), c);
      const float bias = bo[l*DIM + nt*16 + nl];
#pragma unroll
      for (int r = 0; r < 8; ++r) {
        const int m = r + 8*g;
        c[r] += bias + X[m*XS + nt*16 + nl];   // residual (padding rows = 0)
      }
      acc[nt] = c;
    }
    float s1[8], s2[8];
#pragma unroll
    for (int r = 0; r < 8; ++r) {
      float a = 0.f, b2 = 0.f;
#pragma unroll
      for (int nt = 0; nt < 8; ++nt) { const float v = acc[nt][r]; a += v; b2 += v*v; }
#pragma unroll
      for (int off = 1; off < 16; off <<= 1) {
        a  += __shfl_xor(a,  off, 32);
        b2 += __shfl_xor(b2, off, 32);
      }
      s1[r] = a; s2[r] = b2;
    }
#pragma unroll
    for (int nt = 0; nt < 8; ++nt) {
      const float gg = ln_g[l*DIM + nt*16 + nl];
      const float bb = ln_b[l*DIM + nt*16 + nl];
#pragma unroll
      for (int r = 0; r < 8; ++r) {
        const int m = r + 8*g;
        const float mean = s1[r] * (1.f/128.f);
        const float var  = s2[r] * (1.f/128.f) - mean*mean;
        const float inv  = rsqrtf(var + 1e-5f);
        const float y    = (acc[nt][r] - mean)*inv*gg + bb;
        if (m < SEC) X[m*XS + nt*16 + nl] = y;
      }
    }
    lds_fence();
  }

  // ---- write output (coalesced) ----
  {
    float* ob = out + (size_t)batch * SEC * DIM;
    for (int i = lane; i < SEC*DIM; i += 32)
      ob[i] = X[(i >> 7)*XS + (i & 127)];
  }
}

extern "C" void kernel_launch(void* const* d_in, const int* in_sizes, int n_in,
                              void* d_out, int out_size, void* d_ws, size_t ws_size,
                              hipStream_t stream) {
  (void)in_sizes; (void)n_in; (void)out_size; (void)d_ws; (void)ws_size;
  const float* sf     = (const float*)d_in[0];
  const float* regime = (const float*)d_in[1];
  const float* ip_w   = (const float*)d_in[2];
  const float* ip_b   = (const float*)d_in[3];
  const float* gate_w = (const float*)d_in[4];
  const float* gate_b = (const float*)d_in[5];
  const float* wq     = (const float*)d_in[6];
  const float* bq     = (const float*)d_in[7];
  const float* wk     = (const float*)d_in[8];
  const float* bk     = (const float*)d_in[9];
  const float* wv     = (const float*)d_in[10];
  const float* bv     = (const float*)d_in[11];
  const float* wo     = (const float*)d_in[12];
  const float* bo     = (const float*)d_in[13];
  const float* adj    = (const float*)d_in[14];
  const float* ln_g   = (const float*)d_in[15];
  const float* ln_b   = (const float*)d_in[16];
  float* out = (float*)d_out;

  dim3 grid(BATCH / NBAT), block(BLOCK);
  sector_graph_kernel<<<grid, block, 0, stream>>>(
      sf, regime, ip_w, ip_b, gate_w, gate_b, wq, bq, wk, bk, wv, bv,
      wo, bo, adj, ln_g, ln_b, out);
}
